// AttentionHead_65601330479195
// MI455X (gfx1250) — matrix operand order repaired and checked by reading, not compiled, not run on hardware
//
#include <hip/hip_runtime.h>

typedef __bf16 bf16;
typedef bf16  v16bf __attribute__((ext_vector_type(16)));
typedef bf16  v8bf  __attribute__((ext_vector_type(8)));
typedef float v8f   __attribute__((ext_vector_type(8)));
typedef float v4f   __attribute__((ext_vector_type(4)));
typedef int   v4i   __attribute__((ext_vector_type(4)));

#define D_MODEL 1024
#define DIM     128
#define SEQ     4096
#define BATCH   4
#define NE      384   // 3*DIM

#if defined(__gfx1250__) && __has_builtin(__builtin_amdgcn_global_load_async_to_lds_b128) && __has_builtin(__builtin_amdgcn_s_wait_asynccnt)
#define ATTN_ASYNC 1
#else
#define ATTN_ASYNC 0
#endif

static __device__ __forceinline__ v16bf join8(v8bf lo, v8bf hi) {
    return __builtin_shufflevector(lo, hi, 0,1,2,3,4,5,6,7,8,9,10,11,12,13,14,15);
}

static __device__ __forceinline__ v8bf cvt8(v4f a, v4f b) {
    v8bf r;
    r[0]=(bf16)a[0]; r[1]=(bf16)a[1]; r[2]=(bf16)a[2]; r[3]=(bf16)a[3];
    r[4]=(bf16)b[0]; r[5]=(bf16)b[1]; r[6]=(bf16)b[2]; r[7]=(bf16)b[3];
    return r;
}

static __device__ __forceinline__ v8f wmma_bf16(v16bf a, v16bf b, v8f c) {
    // D(f32 16x16) = A(bf16 16x32) * B(bf16 32x16) + C
    return __builtin_amdgcn_wmma_f32_16x16x32_bf16(false, a, false, b, (short)0, c, false, false);
}

#if ATTN_ASYNC
static __device__ __forceinline__ void async_cp16(const bf16* g, bf16* l) {
    __builtin_amdgcn_global_load_async_to_lds_b128(
        (__attribute__((address_space(1))) v4i*)g,
        (__attribute__((address_space(3))) v4i*)l, 0, 0);
}
#endif

// ---------------------------------------------------------------------------
// Kernel 1: W_qkv [1024,384] f32 row-major  ->  Wt [384,1024] bf16 (transposed)
// ---------------------------------------------------------------------------
__global__ void prep_w(const float* __restrict__ W, bf16* __restrict__ Wt) {
    int idx = blockIdx.x * 256 + threadIdx.x;
    if (idx >= D_MODEL * NE) return;
    int k = idx / NE;
    int e = idx - k * NE;
    Wt[e * D_MODEL + k] = (bf16)W[idx];
}

// ---------------------------------------------------------------------------
// Kernel 2: QKV projection.  Each wave: 16 token-rows x 128 output cols
// (grid.y=3 -> x is read only 3x instead of 12x).  De-interleaves columns
// (e%3 -> q/k/v, d=e/3) into Qb, Kb (row-major [T,128]) and Vt (d-major
// [B,128,SEQ]).  Q is pre-scaled by 1/sqrt(128).
// ---------------------------------------------------------------------------
__global__ __launch_bounds__(128) void qkv_proj(
    const float* __restrict__ x, const bf16* __restrict__ Wt,
    const float* __restrict__ bias,
    bf16* __restrict__ Qb, bf16* __restrict__ Kb, bf16* __restrict__ Vt)
{
    const int lane = threadIdx.x & 31;
    const int wave = threadIdx.x >> 5;
    const int half = lane >> 4;
    const int lid  = lane & 15;
    const int rowbase = blockIdx.x * 64 + wave * 16;   // global token row
    const int colbase = blockIdx.y * 128;

    v8f acc[8];
#pragma unroll
    for (int n = 0; n < 8; ++n) acc[n] = (v8f){};

    const float* xrow = x  + (size_t)(rowbase + lid) * D_MODEL + half * 8;
    const bf16*  wcol = Wt + (size_t)(colbase + lid) * D_MODEL + half * 8;

    for (int k0 = 0; k0 < D_MODEL; k0 += 32) {
        v4f a0 = *(const v4f*)(xrow + k0);
        v4f a1 = *(const v4f*)(xrow + k0 + 4);
        v4f a2 = *(const v4f*)(xrow + k0 + 16);
        v4f a3 = *(const v4f*)(xrow + k0 + 20);
        v16bf afrag = join8(cvt8(a0, a1), cvt8(a2, a3));

#pragma unroll
        for (int n = 0; n < 8; ++n) {
            const bf16* wp = wcol + (size_t)n * 16 * D_MODEL + k0;
            v16bf bfrag = join8(*(const v8bf*)wp, *(const v8bf*)(wp + 16));
            acc[n] = wmma_bf16(afrag, bfrag, acc[n]);
        }
    }

    const float qscale = 0.08838834764831845f;  // 128^-0.5 folded into Q
#pragma unroll
    for (int r = 0; r < 8; ++r) {
        int row = rowbase + r + half * 8;
        int t = row & (SEQ - 1);
        int b = row >> 12;           // SEQ == 4096
#pragma unroll
        for (int n = 0; n < 8; ++n) {
            int e = colbase + n * 16 + lid;
            float v = acc[n][r] + bias[e];
            int which = e % 3;
            int d = e / 3;
            if (which == 0)      Qb[(size_t)row * DIM + d] = (bf16)(v * qscale);
            else if (which == 1) Kb[(size_t)row * DIM + d] = (bf16)v;
            else                 Vt[((size_t)b * DIM + d) * SEQ + t] = (bf16)v;
        }
    }
}

// ---------------------------------------------------------------------------
// Block-level staging of one 32-token KV tile into LDS.
// K tile: contiguous 8 KB slice of Kb.  V tile: 128 d-rows x 64 B from Vt.
// Each of the 128 threads issues 8 x 16B transfers (ASYNCcnt += 8 per wave).
// ---------------------------------------------------------------------------
static __device__ __forceinline__ void stage_kv(
    const bf16* __restrict__ kg, const bf16* __restrict__ vg,
    bf16* kl, bf16* vl, int tid)
{
#pragma unroll
    for (int k = 0; k < 4; ++k) {
        int idx = (k * 128 + tid) * 8;
#if ATTN_ASYNC
        async_cp16(kg + idx, kl + idx);
#else
        *(v8bf*)(kl + idx) = *(const v8bf*)(kg + idx);
#endif
    }
    const int c  = (tid & 3) * 8;
    const int d0 = tid >> 2;
#pragma unroll
    for (int k = 0; k < 4; ++k) {
        int d = d0 + k * 32;
        const bf16* g = vg + (size_t)d * SEQ + c;
        bf16* l = vl + d * 32 + c;
#if ATTN_ASYNC
        async_cp16(g, l);
#else
        *(v8bf*)l = *(const v8bf*)g;
#endif
    }
}

// ---------------------------------------------------------------------------
// Kernel 3: fused flash attention.  Block = 4 waves, each wave owns 16 Q rows.
// Double-buffered async KV staging in LDS shared by all 4 waves.  Per 32-wide
// KV tile: S = Q*K^T (8 wmma), online softmax, P*V (8 wmma).
// ---------------------------------------------------------------------------
__global__ __launch_bounds__(128) void attn(
    const bf16* __restrict__ Qb, const bf16* __restrict__ Kb,
    const bf16* __restrict__ Vt, float* __restrict__ out)
{
    __shared__ bf16 Kl[2][32][DIM];    // 2 x 8 KB  (token-major)
    __shared__ bf16 Vl[2][DIM][32];    // 2 x 8 KB  (d-major)
    __shared__ bf16 Plds[4][16][32];   // per-wave P tile

    const int tid   = threadIdx.x;
    const int lane  = tid & 31;
    const int wave  = tid >> 5;
    const int half  = lane >> 4;
    const int lid   = lane & 15;
    const int batch = blockIdx.y;
    const int qglob = batch * SEQ + blockIdx.x * 64 + wave * 16;

    // Q fragments: 16 rows x 128, four K=32 slices, resident in VGPRs
    v16bf qf[4];
    {
        const bf16* qp = Qb + (size_t)(qglob + lid) * DIM + half * 16;
#pragma unroll
        for (int i = 0; i < 4; ++i)
            qf[i] = join8(*(const v8bf*)(qp + i * 32), *(const v8bf*)(qp + i * 32 + 8));
    }

    v8f acc[8];
#pragma unroll
    for (int n = 0; n < 8; ++n) acc[n] = (v8f){};
    float mrow[8], lrow[8];
#pragma unroll
    for (int r = 0; r < 8; ++r) { mrow[r] = -1e30f; lrow[r] = 0.0f; }

    const bf16* kg = Kb + (size_t)batch * SEQ * DIM;   // + kv0*DIM per tile
    const bf16* vg = Vt + (size_t)batch * DIM * SEQ;   // + kv0 per tile
    bf16 (*P)[32] = Plds[wave];

    stage_kv(kg, vg, &Kl[0][0][0], &Vl[0][0][0], tid);

    const int NT = SEQ / 32;
    for (int it = 0; it < NT; ++it) {
        const int cur = it & 1;
        if (it + 1 < NT) {
            stage_kv(kg + (size_t)(it + 1) * 32 * DIM, vg + (it + 1) * 32,
                     &Kl[1 - cur][0][0], &Vl[1 - cur][0][0], tid);
#if ATTN_ASYNC
            __builtin_amdgcn_s_wait_asynccnt(8);   // tile `it` fully in LDS
#endif
        } else {
#if ATTN_ASYNC
            __builtin_amdgcn_s_wait_asynccnt(0);
#endif
        }
        __syncthreads();

        const bf16 (*Kc)[DIM] = Kl[cur];
        const bf16 (*Vc)[32]  = Vl[cur];

        // ---- S = Q * K^T : two 16x16 column sub-tiles ----
        v8f s0 = (v8f){}, s1 = (v8f){};
        const bf16* k0p = &Kc[lid][half * 16];
        const bf16* k1p = &Kc[16 + lid][half * 16];
#pragma unroll
        for (int i = 0; i < 4; ++i) {
            v16bf b0 = join8(*(const v8bf*)(k0p + i * 32), *(const v8bf*)(k0p + i * 32 + 8));
            v16bf b1 = join8(*(const v8bf*)(k1p + i * 32), *(const v8bf*)(k1p + i * 32 + 8));
            s0 = wmma_bf16(qf[i], b0, s0);
            s1 = wmma_bf16(qf[i], b1, s1);
        }

        // ---- online softmax (row stats per lane; reduce across 16 lanes) ----
#pragma unroll
        for (int r = 0; r < 8; ++r) {
            float t = fmaxf(s0[r], s1[r]);
            for (int m = 8; m >= 1; m >>= 1)
                t = fmaxf(t, __shfl_xor(t, m, 32));
            float mnew  = fmaxf(mrow[r], t);
            float alpha = __expf(mrow[r] - mnew);
            float p0 = __expf(s0[r] - mnew);
            float p1 = __expf(s1[r] - mnew);
            float sum = p0 + p1;
            for (int m = 8; m >= 1; m >>= 1)
                sum += __shfl_xor(sum, m, 32);
            lrow[r] = lrow[r] * alpha + sum;
            mrow[r] = mnew;
#pragma unroll
            for (int n = 0; n < 8; ++n) acc[n][r] *= alpha;
            // C-layout -> LDS row-major P[m][kv]
            P[r + half * 8][lid]      = (bf16)p0;
            P[r + half * 8][16 + lid] = (bf16)p1;
        }

        // ---- re-stripe P into an A fragment (same-wave LDS, in-order) ----
        v16bf pf;
        {
            const bf16* pp = &P[lid][half * 16];
            pf = join8(*(const v8bf*)pp, *(const v8bf*)(pp + 8));
        }

        // ---- O += P * V : 8 output d-chunks ----
#pragma unroll
        for (int n = 0; n < 8; ++n) {
            const bf16* vp = &Vc[n * 16 + lid][half * 16];
            v16bf bv = join8(*(const v8bf*)vp, *(const v8bf*)(vp + 8));
            acc[n] = wmma_bf16(pf, bv, acc[n]);
        }
        __syncthreads();
    }

    // ---- epilogue: normalize and store fp32 output [B,SEQ,DIM] ----
#pragma unroll
    for (int r = 0; r < 8; ++r) {
        float inv = 1.0f / lrow[r];
        int row = qglob + r + half * 8;
        float* op = out + (size_t)row * DIM + lid;
#pragma unroll
        for (int n = 0; n < 8; ++n)
            op[n * 16] = acc[n][r] * inv;
    }
}

// ---------------------------------------------------------------------------
extern "C" void kernel_launch(void* const* d_in, const int* in_sizes, int n_in,
                              void* d_out, int out_size, void* d_ws, size_t ws_size,
                              hipStream_t stream) {
    const float* x  = (const float*)d_in[0];   // [4,4096,1024]
    const float* W  = (const float*)d_in[1];   // [1024,384]
    const float* bq = (const float*)d_in[2];   // [384]
    float* out = (float*)d_out;                // [4,4096,128]

    bf16* Qb = (bf16*)d_ws;                                   // [16384,128]  4 MB
    bf16* Kb = Qb + (size_t)BATCH * SEQ * DIM;                // [16384,128]  4 MB
    bf16* Vt = Kb + (size_t)BATCH * SEQ * DIM;                // [4,128,4096] 4 MB
    bf16* Wt = Vt + (size_t)BATCH * SEQ * DIM;                // [384,1024]   0.75 MB

    prep_w<<<dim3((D_MODEL * NE + 255) / 256), dim3(256), 0, stream>>>(W, Wt);
    qkv_proj<<<dim3((BATCH * SEQ) / 64, NE / 128), dim3(128), 0, stream>>>(x, Wt, bq, Qb, Kb, Vt);
    attn<<<dim3(SEQ / 64, BATCH), dim3(128), 0, stream>>>(Qb, Kb, Vt, out);
}
